// LSTMNet_53961969107393
// MI455X (gfx1250) — matrix operand order repaired and checked by reading, not compiled, not run on hardware
//
#include <hip/hip_runtime.h>
#include <hip/hip_bf16.h>

// ---------------------------------------------------------------------------
// Persistent 2-layer LSTM, B=1024, T=1000(+100 future), H=51, on gfx1250.
// One workgroup = 16 batch rows for all 1100 steps. 13 waves, one 16-wide
// gate tile per wave; weight WMMA B-fragments live in VGPRs for the whole
// kernel. GEMMs via v_wmma_f32_16x16x32_f16 (K 51->64, N 204->208).
// Cell state c1/c2 in registers; h1/h2 round-trip through LDS in f16 (WMMA
// A layout); x staged in 128-step LDS chunks (bank-conflict-free stride).
// Fast-path sigmoid/tanh via v_exp_f32 + v_rcp_f32. 5 barriers per step.
// ---------------------------------------------------------------------------

typedef _Float16 h4   __attribute__((ext_vector_type(4)));
typedef _Float16 h8   __attribute__((ext_vector_type(8)));
typedef _Float16 v16h __attribute__((ext_vector_type(16)));
typedef float    f4   __attribute__((ext_vector_type(4)));
typedef float    v8f  __attribute__((ext_vector_type(8)));

#define T_SEQ 1000
#define FUT   100
#define TT    (T_SEQ + FUT)
#define HDIM  51
#define G4    204          // 4*H
#define GP    208          // padded gate count (13 tiles of 16)
#define KP    64           // padded K (2 WMMA chunks of 32)
#define XCH   128          // x staging chunk (steps)
#define XSP   132          // x staging row stride (floats): m*132%64 unique banks
#define NTHR  416          // 13 waves

// LDS layout (bytes), 16B aligned
constexpr int OFF_GB = 0;                      // gates f32 [16][208]     13312
constexpr int OFF_XS = OFF_GB + 16 * GP * 4;   // x chunk f32 [16][132]    8448
constexpr int OFF_H1 = OFF_XS + 16 * XSP * 4;  // h1 f16 [16][64]          2048
constexpr int OFF_H2 = OFF_H1 + 16 * KP * 2;   // h2 f16 [16][64]          2048
constexpr int OFF_OB = OFF_H2 + 16 * KP * 2;   // out feedback f32 [16]      64
constexpr int SMEM_BYTES = OFF_OB + 64;        // 25,920 B

__device__ __forceinline__ float sigm_f(float x) {
    // 1/(1+e^-x) with v_rcp_f32 (no IEEE divide expansion)
    return __builtin_amdgcn_rcpf(1.0f + __expf(-x));
}
__device__ __forceinline__ float tanh_f(float x) {
    // 1 - 2/(e^2x + 1)
    return 1.0f - 2.0f * __builtin_amdgcn_rcpf(__expf(2.0f * x) + 1.0f);
}

// A fragment (16x32 f16) from LDS: lane = M; lanes 0-15: K kb..kb+15
// with kb=0; lanes 16-31: kb=16.  (contiguous K order; B uses the same)
__device__ __forceinline__ v16h load_fragA(const _Float16* rowp, int chunk, int kb) {
    const h8* p0 = (const h8*)(rowp + chunk * 32 + kb);
    const h8* p1 = (const h8*)(rowp + chunk * 32 + kb + 8);
    h8 lo = *p0, hi = *p1;
    return __builtin_shufflevector(lo, hi, 0,1,2,3,4,5,6,7,8,9,10,11,12,13,14,15);
}

// B fragment (32x16 f16) built once from global f32 weights. lane = N = ng;
// lanes 0-15 hold K kb..kb+15 with kb=0, lanes 16-31 with kb=16.
// B[k][n] = W[n][k]  (gates = h @ W^T), zero-padded for k>=51 or n>=204.
__device__ __forceinline__ v16h load_wfragB(const float* __restrict__ W,
                                            int ng, int chunk, int kb) {
    v16h r;
    #pragma unroll
    for (int e = 0; e < 16; ++e) {
        int k = chunk * 32 + kb + e;
        float v = (ng < G4 && k < HDIM) ? W[ng * HDIM + k] : 0.0f;
        r[e] = (_Float16)v;
    }
    return r;
}

__device__ __forceinline__ void lstm_unit(const float* gbuf, _Float16* hbuf,
                                          int m, int j, float& c) {
    float ii = gbuf[m * GP + j];
    float ff = gbuf[m * GP + HDIM + j];
    float gg = gbuf[m * GP + 2 * HDIM + j];
    float oo = gbuf[m * GP + 3 * HDIM + j];
    c = sigm_f(ff) * c + sigm_f(ii) * tanh_f(gg);
    float h = sigm_f(oo) * tanh_f(c);
    hbuf[m * KP + j] = (_Float16)h;
}

__global__ __launch_bounds__(NTHR) void lstm_persistent_kernel(
    const float* __restrict__ x,     // [1024,1000]
    const float* __restrict__ Wih1,  // [204,1]
    const float* __restrict__ Whh1,  // [204,51]
    const float* __restrict__ bih1, const float* __restrict__ bhh1,
    const float* __restrict__ Wih2,  // [204,51]
    const float* __restrict__ Whh2,  // [204,51]
    const float* __restrict__ bih2, const float* __restrict__ bhh2,
    const float* __restrict__ Wl,    // [1,51]
    const float* __restrict__ blp,   // [1]
    float* __restrict__ out)         // [1024,1100]
{
    extern __shared__ char smem[];
    float*    gbuf   = (float*)(smem + OFF_GB);
    float*    xstage = (float*)(smem + OFF_XS);
    _Float16* h1     = (_Float16*)(smem + OFF_H1);
    _Float16* h2     = (_Float16*)(smem + OFF_H2);
    float*    obuf   = (float*)(smem + OFF_OB);

    const int tid  = threadIdx.x;
    const int lane = tid & 31;
    const int wave = tid >> 5;          // 0..12 == gate tile index
    const int b0   = blockIdx.x * 16;   // batch tile base

    const int arow = lane & 15;                 // M (A) / N (B) within tile
    const int kb_a = (lane < 16) ? 0 : 16;      // A-fragment K base
    const int kb_b = (lane < 16) ? 0 : 16;      // B-fragment K base
    const int mofs = (lane < 16) ? 0 : 8;       // C/D: m = r + mofs
    const int ng   = wave * 16 + arow;          // global gate column (0..207)

    // ---- one-time: weights -> VGPR-resident f16 WMMA B fragments ----
    const v16h b1c0  = load_wfragB(Whh1, ng, 0, kb_b);
    const v16h b1c1  = load_wfragB(Whh1, ng, 1, kb_b);
    const v16h b2ic0 = load_wfragB(Wih2, ng, 0, kb_b);
    const v16h b2ic1 = load_wfragB(Wih2, ng, 1, kb_b);
    const v16h b2hc0 = load_wfragB(Whh2, ng, 0, kb_b);
    const v16h b2hc1 = load_wfragB(Whh2, ng, 1, kb_b);
    const float wv     = (ng < G4) ? Wih1[ng] : 0.0f;            // IN==1 rank-1 term
    const float bias1v = (ng < G4) ? (bih1[ng] + bhh1[ng]) : 0.0f;
    const float bias2v = (ng < G4) ? (bih2[ng] + bhh2[ng]) : 0.0f;
    const float blv    = blp[0];

    // output-projection taps (threads 0..255: row frow, 4 taps at fsub*4)
    const int frow = tid >> 4;
    const int fsub = tid & 15;
    float wlv[4];
    #pragma unroll
    for (int e = 0; e < 4; ++e) {
        int k = fsub * 4 + e;
        wlv[e] = (tid < 256 && k < HDIM) ? Wl[k] : 0.0f;
    }

    // ---- zero state: h in LDS (incl. K padding), c in registers ----
    for (int i = tid; i < 16 * KP; i += NTHR) { h1[i] = (_Float16)0.0f; h2[i] = (_Float16)0.0f; }
    float c1a = 0.0f, c1b = 0.0f, c2a = 0.0f, c2b = 0.0f;   // units tid, tid+416
    __syncthreads();

    for (int t = 0; t < TT; ++t) {
        // ---- stage 128 steps of x into LDS (one HBM trip / 128 steps) ----
        if ((t & (XCH - 1)) == 0 && t < T_SEQ) {
            for (int idx = tid; idx < 16 * XCH; idx += NTHR) {
                int m = idx >> 7, c = idx & (XCH - 1);
                int tt = t + c;
                xstage[m * XSP + c] = (tt < T_SEQ) ? x[(size_t)(b0 + m) * T_SEQ + tt] : 0.0f;
            }
            __syncthreads();
        }

        // ---- phase B: layer-1 gates = bias1 + x*Wih1 + h1 @ Whh1^T ----
        // (also prefetch h2 A-fragments: h2 is stable until phase E)
        v16h a20 = load_fragA(h2 + arow * KP, 0, kb_a);
        v16h a21 = load_fragA(h2 + arow * KP, 1, kb_a);
        {
            v16h a0 = load_fragA(h1 + arow * KP, 0, kb_a);
            v16h a1 = load_fragA(h1 + arow * KP, 1, kb_a);
            float xr[8];
            if (t < T_SEQ) {                       // uniform branch
                int col = t & (XCH - 1);
                #pragma unroll
                for (int r = 0; r < 8; ++r)
                    xr[r] = xstage[(r + mofs) * XSP + col];   // distinct banks per m
            } else {
                f4 xa = *(const f4*)(obuf + mofs);
                f4 xb = *(const f4*)(obuf + mofs + 4);
                #pragma unroll
                for (int r = 0; r < 4; ++r) { xr[r] = xa[r]; xr[r + 4] = xb[r]; }
            }
            v8f acc;
            #pragma unroll
            for (int r = 0; r < 8; ++r)
                acc[r] = bias1v + xr[r] * wv;
            acc = __builtin_amdgcn_wmma_f32_16x16x32_f16(false, a0, false, b1c0, (short)0, acc, false, false);
            acc = __builtin_amdgcn_wmma_f32_16x16x32_f16(false, a1, false, b1c1, (short)0, acc, false, false);
            #pragma unroll
            for (int r = 0; r < 8; ++r)
                gbuf[(r + mofs) * GP + ng] = acc[r];
        }
        __syncthreads();

        // ---- phase C: layer-1 elementwise (c1 in regs, h1 -> LDS f16) ----
        {
            int m = tid / HDIM, j = tid - m * HDIM;           // unit tid  (<416)
            lstm_unit(gbuf, h1, m, j, c1a);
            int u2 = tid + NTHR;                              // unit tid+416
            if (u2 < 16 * HDIM) {
                int m2 = u2 / HDIM, j2 = u2 - m2 * HDIM;
                lstm_unit(gbuf, h1, m2, j2, c1b);
            }
        }
        __syncthreads();

        // ---- phase D: layer-2 gates = bias2 + h1 @ Wih2^T + h2 @ Whh2^T ----
        // two independent 2-deep WMMA chains to hide D->C RAW hazard gaps
        {
            v16h a10 = load_fragA(h1 + arow * KP, 0, kb_a);
            v16h a11 = load_fragA(h1 + arow * KP, 1, kb_a);
            v8f accA, accB;
            #pragma unroll
            for (int r = 0; r < 8; ++r) { accA[r] = bias2v; accB[r] = 0.0f; }
            accA = __builtin_amdgcn_wmma_f32_16x16x32_f16(false, a10, false, b2ic0, (short)0, accA, false, false);
            accB = __builtin_amdgcn_wmma_f32_16x16x32_f16(false, a20, false, b2hc0, (short)0, accB, false, false);
            accA = __builtin_amdgcn_wmma_f32_16x16x32_f16(false, a11, false, b2ic1, (short)0, accA, false, false);
            accB = __builtin_amdgcn_wmma_f32_16x16x32_f16(false, a21, false, b2hc1, (short)0, accB, false, false);
            #pragma unroll
            for (int r = 0; r < 8; ++r)
                gbuf[(r + mofs) * GP + ng] = accA[r] + accB[r];
        }
        __syncthreads();

        // ---- phase E: layer-2 elementwise (c2 in regs, h2 -> LDS f16) ----
        {
            int m = tid / HDIM, j = tid - m * HDIM;
            lstm_unit(gbuf, h2, m, j, c2a);
            int u2 = tid + NTHR;
            if (u2 < 16 * HDIM) {
                int m2 = u2 / HDIM, j2 = u2 - m2 * HDIM;
                lstm_unit(gbuf, h2, m2, j2, c2b);
            }
        }
        __syncthreads();

        // ---- phase F: out = h2 @ Wl^T + bl (16 lanes/row, shfl butterfly) ----
        if (tid < 256) {
            h4 hv = *(const h4*)(h2 + frow * KP + fsub * 4);  // zero-padded row
            float s = (float)hv[0] * wlv[0] + (float)hv[1] * wlv[1]
                    + (float)hv[2] * wlv[2] + (float)hv[3] * wlv[3];
            s += __shfl_xor(s, 1, 32);
            s += __shfl_xor(s, 2, 32);
            s += __shfl_xor(s, 4, 32);
            s += __shfl_xor(s, 8, 32);
            if (fsub == 0) {
                float o = s + blv;
                obuf[frow] = o;
                out[(size_t)(b0 + frow) * TT + t] = o;
            }
        }
        __syncthreads();
    }
}

extern "C" void kernel_launch(void* const* d_in, const int* in_sizes, int n_in,
                              void* d_out, int out_size, void* d_ws, size_t ws_size,
                              hipStream_t stream) {
    const float* x    = (const float*)d_in[0];
    const float* Wih1 = (const float*)d_in[1];
    const float* Whh1 = (const float*)d_in[2];
    const float* bih1 = (const float*)d_in[3];
    const float* bhh1 = (const float*)d_in[4];
    const float* Wih2 = (const float*)d_in[5];
    const float* Whh2 = (const float*)d_in[6];
    const float* bih2 = (const float*)d_in[7];
    const float* bhh2 = (const float*)d_in[8];
    const float* Wl   = (const float*)d_in[9];
    const float* bl   = (const float*)d_in[10];
    float* out = (float*)d_out;

    // 64 workgroups (1024/16 batch tiles) x 13 waves; ~25 KB dynamic LDS each.
    lstm_persistent_kernel<<<dim3(64), dim3(NTHR), SMEM_BYTES, stream>>>(
        x, Wih1, Whh1, bih1, bhh1, Wih2, Whh2, bih2, bhh2, Wl, bl, out);
}
